// ARAttention_40767829574370
// MI455X (gfx1250) — matrix operand order repaired and checked by reading, not compiled, not run on hardware
//
#include <hip/hip_runtime.h>
#include <hip/hip_bf16.h>

typedef __attribute__((ext_vector_type(16))) _Float16     v16h;
typedef __attribute__((ext_vector_type(8)))  float        v8f;
typedef __attribute__((ext_vector_type(4)))  unsigned int v4u;
typedef __attribute__((ext_vector_type(8)))  int          v8i;
typedef __attribute__((ext_vector_type(4)))  int          v4i;

#define N_B    32
#define T_LEN  4096
#define ENC_HD 512
#define ATT_HD 256
#define DEC_HD 512
#define OUT_D  80
#define SPK_D  64
#define KWID   31
#define ATT_R  10

// workspace byte offsets (all 32B-aligned)
#define WS_BIAS   0u          // N*256 f32   = 32 KB
#define WS_WIN    32768u      // N*2  int    = 256 B
#define WS_LOGIT  65536u      // N*T  f32    = 512 KB
#define WS_BHI    655360u     // 272*512 f16 = 272 KB
#define WS_BLO    983040u     // 272*512 f16 = 272 KB

#if defined(__has_builtin)
#if __has_builtin(__builtin_amdgcn_tensor_load_to_lds) && __has_builtin(__builtin_amdgcn_s_wait_tensorcnt)
#define HAVE_TDM 1
#endif
#endif

// ---------------------------------------------------------------------------
// Kernel 1: per-batch setup: prenet GEMVs, per-batch bias vector, argmax window
// ---------------------------------------------------------------------------
__global__ void setup_kernel(const float* __restrict__ input_dec,
                             const float* __restrict__ prev_att,
                             const float* __restrict__ spkr_vec,
                             const int*   __restrict__ lengths,
                             const float* __restrict__ speed,
                             const float* __restrict__ W_spkr,
                             const float* __restrict__ W_dec,
                             const float* __restrict__ W_speed,
                             const float* __restrict__ Wp1, const float* __restrict__ bp1,
                             const float* __restrict__ Wp2, const float* __restrict__ bp2,
                             float* __restrict__ bias_out, int* __restrict__ win_out) {
  __shared__ float s_in[OUT_D + SPK_D];
  __shared__ float s_h1[2 * DEC_HD];
  __shared__ float s_h2[DEC_HD];
  __shared__ float s_val[256];
  __shared__ int   s_idx[256];
  const int n = blockIdx.x, tid = threadIdx.x;

  if (tid < OUT_D + SPK_D)
    s_in[tid] = (tid < OUT_D) ? input_dec[n * OUT_D + tid]
                              : spkr_vec[n * SPK_D + (tid - OUT_D)];
  __syncthreads();

  for (int j = tid; j < 2 * DEC_HD; j += 256) {
    float s = bp1[j];
    for (int i = 0; i < OUT_D + SPK_D; ++i) s += s_in[i] * Wp1[i * (2 * DEC_HD) + j];
    s_h1[j] = fmaxf(s, 0.f);
  }
  __syncthreads();
  for (int j = tid; j < DEC_HD; j += 256) {
    float s = bp2[j];
    for (int i = 0; i < 2 * DEC_HD; ++i) s += s_h1[i] * Wp2[i * DEC_HD + j];
    s_h2[j] = fmaxf(s, 0.f);
  }
  __syncthreads();
  {
    const int c = tid;  // 256 threads == ATT_HD columns
    float dec_att = 0.f;
    for (int i = 0; i < DEC_HD; ++i) dec_att += s_h2[i] * W_dec[i * ATT_HD + c];
    float sp = 0.f;
    for (int i = 0; i < SPK_D; ++i) sp += spkr_vec[n * SPK_D + i] * W_spkr[i * ATT_HD + c];
    sp = sp / (1.f + fabsf(sp));                  // softsign
    const float spd = speed[n] * W_speed[c];
    bias_out[n * ATT_HD + c] = dec_att + sp + spd;
  }
  // argmax of prev attention (first-index tie break, matching jnp.argmax)
  float best = -1.f; int bidx = 0;
  for (int t = tid; t < T_LEN; t += 256) {
    const float v = prev_att[(size_t)n * T_LEN + t];
    if (v > best) { best = v; bidx = t; }
  }
  s_val[tid] = best; s_idx[tid] = bidx;
  __syncthreads();
  for (int s = 128; s > 0; s >>= 1) {
    if (tid < s) {
      const float v2 = s_val[tid + s]; const int i2 = s_idx[tid + s];
      if (v2 > s_val[tid] || (v2 == s_val[tid] && i2 < s_idx[tid])) {
        s_val[tid] = v2; s_idx[tid] = i2;
      }
    }
    __syncthreads();
  }
  if (tid == 0) {
    const int mi = s_idx[0];
    win_out[n * 2 + 0] = max(mi - (ATT_R - 1), 0);
    win_out[n * 2 + 1] = min(mi + (ATT_R - 1), lengths[n] - 1);
  }
}

// ---------------------------------------------------------------------------
// Kernel 2: pack W_enc (chunks 0..15) and conv_w im2col B (chunk 16) into
// WMMA B-operand lane order, split into hi/lo f16.
// B layout (16-bit, 32x16): lane = 16*h + nn holds K = ((idx&8)<<1) + 8*h + (idx&7), N = nn.
// ---------------------------------------------------------------------------
__global__ void pack_B(const float* __restrict__ W_enc,
                       const float* __restrict__ conv_w,
                       _Float16* __restrict__ Bhi, _Float16* __restrict__ Blo) {
  const int tile = blockIdx.x;          // chunk*16 + ct,  0..271
  const int chunk = tile >> 4, ct = tile & 15;
  const int e = threadIdx.x;            // 0..511  (= lane*16 + idx)
  const int L = e >> 4, idx = e & 15;
  const int h = L >> 4, nn = L & 15;
  const int klocal = ((idx & 8) << 1) + 8 * h + (idx & 7);
  const int c = ct * 16 + nn;
  float val;
  if (chunk < 16) {
    val = W_enc[(chunk * 32 + klocal) * ATT_HD + c];
  } else {
    val = (klocal < KWID) ? conv_w[c * KWID + klocal] : 0.f;
  }
  const _Float16 hi = (_Float16)val;
  const _Float16 lo = (_Float16)(val - (float)hi);
  Bhi[(size_t)tile * 512 + e] = hi;
  Blo[(size_t)tile * 512 + e] = lo;
}

// ---------------------------------------------------------------------------
// TDM: DMA one contiguous 16 KB block (2048 x 8B elements) global -> LDS.
// D# built per CDNA5 ISA ch.8: group0 {count=1, lds_addr, global_addr, type=2},
// group1 {wg_mask=0, data_size=8B, tensor_dim0=tile_dim0=2048}, dims 2..4 = 1.
// ---------------------------------------------------------------------------
#ifdef HAVE_TDM
__device__ __forceinline__ void tdm_copy16k(uint64_t gsrc, uint32_t lds_byte) {
  v4u g0;
  g0[0] = 1u;                                   // count=1, no gather, user mode
  g0[1] = lds_byte;                             // lds_addr
  g0[2] = (unsigned int)gsrc;                   // global_addr[31:0]
  g0[3] = (unsigned int)(gsrc >> 32) | (2u << 30);  // global_addr[56:32] | type=2
  v8i g1;
  g1[0] = 3 << 16;                              // workgroup_mask=0, data_size=8B
  g1[1] = (2048 & 0xFFFF) << 16;                // tensor_dim0[15:0] @ bits 63:48
  g1[2] = (2048 >> 16) | (1 << 16);             // tensor_dim0[31:16], tensor_dim1=1
  g1[3] = (2048 << 16);                         // tile_dim0 = 2048 @ bits 127:112
  g1[4] = 0;                                    // tile_dim1=0 (1D), tile_dim2=0
  g1[5] = 2048;                                 // tensor_dim0_stride lo32
  g1[6] = 0;
  g1[7] = 0;
  v4i g2; g2[0] = 1; g2[1] = 1; g2[2] = 0; g2[3] = 0;   // tensor_dim2=1, tensor_dim3=1
  v4i g3; g3[0] = 0; g3[1] = 1 << 16; g3[2] = 0; g3[3] = 0;  // tensor_dim4=1
#if __clang_major__ >= 23
  v8i g4 = (v8i)0;
  __builtin_amdgcn_tensor_load_to_lds(g0, g1, g2, g3, g4, 0);
#else
  __builtin_amdgcn_tensor_load_to_lds(g0, g1, g2, g3, 0);
#endif
}
#endif

// ---------------------------------------------------------------------------
// Kernel 3: fused  logit = tanh(softsign(enc@W_enc + b_enc) + conv + bias) @ W_proj
// Block: one batch n, 128 rows. 8 waves x (16 rows x 256 cols) = 16 acc tiles/wave.
// B operands staged global->LDS by the Tensor Data Mover, double-buffered, so
// each 32 KB chunk is fetched once per block instead of once per wave.
// hi/lo f16 split (3 WMMAs per chunk per tile) -> fp32-level accuracy.
// ---------------------------------------------------------------------------
__global__ void __launch_bounds__(256)
attn_main(const float* __restrict__ input_enc,
          const float* __restrict__ prev_att,
          const float* __restrict__ b_enc,
          const float* __restrict__ W_proj,
          const float* __restrict__ b_proj,
          const float* __restrict__ bias_n,
          const _Float16* __restrict__ Bhi, const _Float16* __restrict__ Blo,
          float* __restrict__ logit_out) {
  __shared__ __align__(32) char s_B[2][32768];   // [buf][hi 16K | lo 16K]
  __shared__ float s_benc[ATT_HD], s_bias[ATT_HD], s_wproj[ATT_HD];
  const int n   = blockIdx.y;
  const int t0  = blockIdx.x * 128;
  const int tid = threadIdx.x;
  const int wave = tid >> 5;
  const int lane = tid & 31;
  const int h = lane >> 4;      // K-half for A/B operands
  const int m = lane & 15;      // A row within tile / C column

  if (tid < ATT_HD) {
    s_benc[tid]  = b_enc[tid];
    s_bias[tid]  = bias_n[n * ATT_HD + tid];
    s_wproj[tid] = W_proj[tid];
  }

  const int t = t0 + wave * 16 + m;                       // this lane's A row
  const float* arow = input_enc + (size_t)(n * T_LEN + t) * ENC_HD;

  v8f zero = {};
  v8f acc[16];
#pragma unroll
  for (int i = 0; i < 16; ++i) acc[i] = zero;

  // ---- B chunk staging (double buffered) ----
#ifdef HAVE_TDM
  #define STAGE_B(chunkv, bufv)                                                   \
    do { if (wave == 0) {                                                         \
      tdm_copy16k((uint64_t)(uintptr_t)(Bhi + (size_t)(chunkv) * 16 * 512),       \
                  (uint32_t)(uintptr_t)&s_B[(bufv)][0]);                          \
      tdm_copy16k((uint64_t)(uintptr_t)(Blo + (size_t)(chunkv) * 16 * 512),       \
                  (uint32_t)(uintptr_t)&s_B[(bufv)][16384]);                      \
    } } while (0)
  #define WAIT_B() do { if (wave == 0) __builtin_amdgcn_s_wait_tensorcnt(0); } while (0)
#else
  // fallback: cooperative vector copy (256 threads x 8 x 16B = 32 KB)
  #define STAGE_B(chunkv, bufv)                                                   \
    do {                                                                          \
      const float4* __sh = (const float4*)(Bhi + (size_t)(chunkv) * 16 * 512);    \
      const float4* __sl = (const float4*)(Blo + (size_t)(chunkv) * 16 * 512);    \
      float4* __d = (float4*)&s_B[(bufv)][0];                                     \
      _Pragma("unroll")                                                           \
      for (int r = 0; r < 4; ++r) __d[r * 256 + tid] = __sh[r * 256 + tid];       \
      _Pragma("unroll")                                                           \
      for (int r = 0; r < 4; ++r) __d[1024 + r * 256 + tid] = __sl[r * 256 + tid];\
    } while (0)
  #define WAIT_B() do { } while (0)
#endif

  STAGE_B(0, 0);
  int cur = 0;

  for (int chunk = 0; chunk < 16; ++chunk) {
    WAIT_B();
    __syncthreads();                      // buf[cur] ready; buf[cur^1] free
    STAGE_B(chunk + 1, cur ^ 1);          // prefetch next (incl. conv chunk 16)

    // A operand: lane(h,m), element idx -> K = (idx&8)*2 + 8h + (idx&7)
    // => two contiguous 8-float runs at k = 8h and k = 16+8h within the chunk.
    const float4* p0 = reinterpret_cast<const float4*>(arow + chunk * 32 + 8 * h);
    const float4* p1 = reinterpret_cast<const float4*>(arow + chunk * 32 + 16 + 8 * h);
    const float4 x0 = p0[0], x1 = p0[1], y0 = p1[0], y1 = p1[1];
    const float av[16] = {x0.x, x0.y, x0.z, x0.w, x1.x, x1.y, x1.z, x1.w,
                          y0.x, y0.y, y0.z, y0.w, y1.x, y1.y, y1.z, y1.w};
    v16h a_hi, a_lo;
#pragma unroll
    for (int i = 0; i < 16; ++i) {
      const _Float16 hv = (_Float16)av[i];
      a_hi[i] = hv;
      a_lo[i] = (_Float16)(av[i] - (float)hv);
    }
    const char* bufp = &s_B[cur][0];
#pragma unroll
    for (int ct = 0; ct < 16; ++ct) {
      const v16h b_hi = *reinterpret_cast<const v16h*>(bufp + ct * 1024 + lane * 32);
      const v16h b_lo = *reinterpret_cast<const v16h*>(bufp + 16384 + ct * 1024 + lane * 32);
      acc[ct] = __builtin_amdgcn_wmma_f32_16x16x32_f16(false, a_hi, false, b_hi, (short)0, acc[ct], false, false);
      acc[ct] = __builtin_amdgcn_wmma_f32_16x16x32_f16(false, a_hi, false, b_lo, (short)0, acc[ct], false, false);
      acc[ct] = __builtin_amdgcn_wmma_f32_16x16x32_f16(false, a_lo, false, b_hi, (short)0, acc[ct], false, false);
    }
    cur ^= 1;
  }

  // Location-conv A operand: im2col of prev_attention, K=32 (31 taps + pad)
  v16h ca_hi, ca_lo;
#pragma unroll
  for (int i = 0; i < 16; ++i) {
    const int klocal = ((i & 8) << 1) + 8 * h + (i & 7);
    const int pos = t - 15 + klocal;
    float v = 0.f;
    if (klocal < KWID && pos >= 0 && pos < T_LEN) v = prev_att[(size_t)n * T_LEN + pos];
    const _Float16 hv = (_Float16)v;
    ca_hi[i] = hv;
    ca_lo[i] = (_Float16)(v - (float)hv);
  }

  WAIT_B();
  __syncthreads();                        // conv-chunk B ready in buf[cur]

  const float bp = b_proj[0];
  float part[8];
#pragma unroll
  for (int r = 0; r < 8; ++r) part[r] = 0.f;

  const char* cbufp = &s_B[cur][0];
#pragma unroll
  for (int ct = 0; ct < 16; ++ct) {
    const v16h b_hi = *reinterpret_cast<const v16h*>(cbufp + ct * 1024 + lane * 32);
    const v16h b_lo = *reinterpret_cast<const v16h*>(cbufp + 16384 + ct * 1024 + lane * 32);
    v8f conv = zero;
    conv = __builtin_amdgcn_wmma_f32_16x16x32_f16(false, ca_hi, false, b_hi, (short)0, conv, false, false);
    conv = __builtin_amdgcn_wmma_f32_16x16x32_f16(false, ca_hi, false, b_lo, (short)0, conv, false, false);
    conv = __builtin_amdgcn_wmma_f32_16x16x32_f16(false, ca_lo, false, b_hi, (short)0, conv, false, false);
    const int c = ct * 16 + m;        // this lane's column (C/D: N = lane%16)
    const float be = s_benc[c], bn = s_bias[c], wp = s_wproj[c];
#pragma unroll
    for (int r = 0; r < 8; ++r) {     // C/D: row = r + 8*h
      const float g = acc[ct][r] + be;
      const float ssg = g / (1.f + fabsf(g));       // softsign
      const float e = ssg + conv[r] + bn;
      part[r] += tanhf(e) * wp;
    }
  }
  // reduce over the 16 lanes of this half (columns) -> full 256-col dot
#pragma unroll
  for (int mask = 1; mask <= 8; mask <<= 1) {
#pragma unroll
    for (int r = 0; r < 8; ++r) part[r] += __shfl_xor(part[r], mask, 32);
  }
  if (m == 0) {
    const int row = t0 + wave * 16 + 8 * h;
#pragma unroll
    for (int r = 0; r < 8; ++r)
      logit_out[(size_t)n * T_LEN + row + r] = part[r] + bp;
  }
}

// ---------------------------------------------------------------------------
// Kernel 4: global max over T, masked-window softmax, zeros elsewhere
// ---------------------------------------------------------------------------
__global__ void finalize_kernel(const float* __restrict__ logit,
                                const int* __restrict__ win,
                                float* __restrict__ out) {
  __shared__ float red[256];
  const int n = blockIdx.x, tid = threadIdx.x;
  const float* lg = logit + (size_t)n * T_LEN;
  const int lo = win[n * 2 + 0], hi = win[n * 2 + 1];

  float mx = -3.4e38f;
  for (int t = tid; t < T_LEN; t += 256) mx = fmaxf(mx, lg[t]);
  red[tid] = mx; __syncthreads();
  for (int s = 128; s > 0; s >>= 1) {
    if (tid < s) red[tid] = fmaxf(red[tid], red[tid + s]);
    __syncthreads();
  }
  const float M = red[0];
  __syncthreads();

  float sm = 0.f;
  for (int t = tid; t < T_LEN; t += 256)
    if (t >= lo && t <= hi) sm += expf(lg[t] - M);
  red[tid] = sm; __syncthreads();
  for (int s = 128; s > 0; s >>= 1) {
    if (tid < s) red[tid] += red[tid + s];
    __syncthreads();
  }
  const float denom = fmaxf(red[0], 1e-12f);

  for (int t = tid; t < T_LEN; t += 256) {
    float v = 0.f;
    if (t >= lo && t <= hi) v = expf(lg[t] - M) / denom;
    out[(size_t)n * T_LEN + t] = v;
  }
}

// ---------------------------------------------------------------------------
extern "C" void kernel_launch(void* const* d_in, const int* in_sizes, int n_in,
                              void* d_out, int out_size, void* d_ws, size_t ws_size,
                              hipStream_t stream) {
  (void)in_sizes; (void)n_in; (void)out_size; (void)ws_size;
  const float* input_enc = (const float*)d_in[0];
  const float* input_dec = (const float*)d_in[1];
  const float* prev_att  = (const float*)d_in[2];
  const float* spkr_vec  = (const float*)d_in[3];
  const int*   lengths   = (const int*)  d_in[4];
  const float* speed     = (const float*)d_in[5];
  const float* W_enc     = (const float*)d_in[6];
  const float* b_enc     = (const float*)d_in[7];
  const float* W_spkr    = (const float*)d_in[8];
  const float* conv_w    = (const float*)d_in[9];
  const float* W_dec     = (const float*)d_in[10];
  const float* W_speed   = (const float*)d_in[11];
  const float* Wp1       = (const float*)d_in[12];
  const float* bp1       = (const float*)d_in[13];
  const float* Wp2       = (const float*)d_in[14];
  const float* bp2       = (const float*)d_in[15];
  const float* W_proj    = (const float*)d_in[16];
  const float* b_proj    = (const float*)d_in[17];

  char* ws = (char*)d_ws;
  float*    ws_bias  = (float*)(ws + WS_BIAS);
  int*      ws_win   = (int*)  (ws + WS_WIN);
  float*    ws_logit = (float*)(ws + WS_LOGIT);
  _Float16* ws_bhi   = (_Float16*)(ws + WS_BHI);
  _Float16* ws_blo   = (_Float16*)(ws + WS_BLO);

  setup_kernel<<<N_B, 256, 0, stream>>>(input_dec, prev_att, spkr_vec, lengths, speed,
                                        W_spkr, W_dec, W_speed, Wp1, bp1, Wp2, bp2,
                                        ws_bias, ws_win);
  pack_B<<<272, 512, 0, stream>>>(W_enc, conv_w, ws_bhi, ws_blo);
  attn_main<<<dim3(T_LEN / 128, N_B), 256, 0, stream>>>(input_enc, prev_att, b_enc,
                                                        W_proj, b_proj, ws_bias,
                                                        ws_bhi, ws_blo, ws_logit);
  finalize_kernel<<<N_B, 256, 0, stream>>>(ws_logit, ws_win, (float*)d_out);
}
